// MultiHeadAttention_22668837388405
// MI455X (gfx1250) — matrix operand order repaired and checked by reading, not compiled, not run on hardware
//
#include <hip/hip_runtime.h>
#include <cmath>

// ---------------------------------------------------------------------------
// MultiHeadAttention for MI455X (gfx1250), wave32 + WMMA f32_16x16x32_f16.
// K tiles staged via TDM (tensor_load_to_lds, TENSORcnt); V tiles staged via
// global_load_async_to_lds_b128 (ASYNCcnt); both double-buffered in LDS.
//   B=4, S=2048, D_MODEL=256, H=8, HEAD_DIM=32
// ---------------------------------------------------------------------------

#define DM 256
#define NH 8
#define HD 32
#define BB 4
#define SS 2048

typedef __attribute__((ext_vector_type(16))) _Float16     v16h;
typedef __attribute__((ext_vector_type(8)))  float        v8f;
typedef __attribute__((ext_vector_type(4)))  unsigned int u32x4;
typedef __attribute__((ext_vector_type(8)))  int          i32x8;
typedef __attribute__((ext_vector_type(4)))  int          i32x4;

#if __has_builtin(__builtin_amdgcn_tensor_load_to_lds)
#define USE_TDM 1
#else
#define USE_TDM 0
#endif

union HFrag {
    v16h     v;
    _Float16 e[16];
    u32x4    q[2];
};

static __device__ __forceinline__ v8f wmma_f16(v16h a, v16h b, v8f c) {
    // (neg_a, A, neg_b, B, c_mod, C, reuse_a, reuse_b)
    return __builtin_amdgcn_wmma_f32_16x16x32_f16(false, a, false, b,
                                                  (short)0, c, false, false);
}

static __device__ __forceinline__ float rowmax16(float v) {
    #pragma unroll
    for (int off = 1; off < 16; off <<= 1)
        v = fmaxf(v, __shfl_xor(v, off, 32));
    return v;
}
static __device__ __forceinline__ float rowsum16(float v) {
    #pragma unroll
    for (int off = 1; off < 16; off <<= 1)
        v += __shfl_xor(v, off, 32);
    return v;
}

// --- CDNA5 async memory->LDS (ASYNCcnt) ------------------------------------
static __device__ __forceinline__ void async_copy_b128(unsigned lds_byte,
                                                       const void* gptr) {
    unsigned long long ga = (unsigned long long)(uintptr_t)gptr;
    // VDST = LDS byte address VGPR, VADDR = 64-bit global address (GV mode)
    asm volatile("global_load_async_to_lds_b128 %0, %1, off"
                 :: "v"(lds_byte), "v"(ga)
                 : "memory");
}

static __device__ __forceinline__ void wait_async_le4() {
#if __has_builtin(__builtin_amdgcn_s_wait_asynccnt)
    __builtin_amdgcn_s_wait_asynccnt(4);
#else
    asm volatile("s_wait_asynccnt 0x4" ::: "memory");
#endif
    asm volatile("" ::: "memory");
}
static __device__ __forceinline__ void wait_async_le8() {
#if __has_builtin(__builtin_amdgcn_s_wait_asynccnt)
    __builtin_amdgcn_s_wait_asynccnt(8);
#else
    asm volatile("s_wait_asynccnt 0x8" ::: "memory");
#endif
    asm volatile("" ::: "memory");
}
static __device__ __forceinline__ void wait_async_0() {
#if __has_builtin(__builtin_amdgcn_s_wait_asynccnt)
    __builtin_amdgcn_s_wait_asynccnt(0);
#else
    asm volatile("s_wait_asynccnt 0x0" ::: "memory");
#endif
    asm volatile("" ::: "memory");
}
static __device__ __forceinline__ void wait_dscnt_0() {
    asm volatile("s_wait_dscnt 0x0" ::: "memory");
}

// --- CDNA5 Tensor Data Mover (TENSORcnt) -----------------------------------
#if USE_TDM
// 2-D tile load: 32 (dim1 rows) x 32 (dim0 elems) of 2-byte data into LDS,
// rows compacted (tile rows stored back-to-back, 64B each).
// D# packing per ISA 8.3/8.4: group0 = {count/type/lds_addr/global_addr},
// group1 = {data_size, tensor_dim0/1, tile_dim0/1/2, tensor_dim0_stride}.
static __device__ __forceinline__ void tdm_load_tile32x32_f16(
        unsigned lds_byte, const void* gaddr,
        unsigned tensor_d0, unsigned tensor_d1, unsigned stride_d0) {
    unsigned long long ga = (unsigned long long)(uintptr_t)gaddr;
    u32x4 g0;
    g0[0] = 1u;                                       // count=1 (valid user D#)
    g0[1] = lds_byte;                                 // lds_addr
    g0[2] = (unsigned)(ga & 0xffffffffu);             // global_addr[31:0]
    g0[3] = (unsigned)((ga >> 32) & 0x01ffffffu)      // global_addr[56:32]
          | (2u << 30);                               // type=2 ("image")
    i32x8 g1;
    g1[0] = 0x00010000;                               // data_size=1 (2 bytes)
    g1[1] = (int)((tensor_d0 & 0xffffu) << 16);       // tensor_dim0[15:0]
    g1[2] = (int)(((tensor_d0 >> 16) & 0xffffu)       // tensor_dim0[31:16]
          |       ((tensor_d1 & 0xffffu) << 16));     // tensor_dim1[15:0]
    g1[3] = (int)(((tensor_d1 >> 16) & 0xffffu)       // tensor_dim1[31:16]
          |       (32u << 16));                       // tile_dim0 = 32
    g1[4] = 32;                                       // tile_dim1=32, tile_dim2=0
    g1[5] = (int)stride_d0;                           // tensor_dim0_stride[31:0]
    g1[6] = 0;                                        // stride hi / dim1_stride lo
    g1[7] = 0;
    const i32x4 z4 = {0, 0, 0, 0};
#if __clang_major__ >= 23
    const i32x8 z8 = {0, 0, 0, 0, 0, 0, 0, 0};
    __builtin_amdgcn_tensor_load_to_lds(g0, g1, z4, z4, z8, 0);
#else
    __builtin_amdgcn_tensor_load_to_lds(g0, g1, z4, z4, 0);
#endif
}
static __device__ __forceinline__ void wait_tensor_le1() {
    __builtin_amdgcn_s_wait_tensorcnt(1);
    asm volatile("" ::: "memory");
}
static __device__ __forceinline__ void wait_tensor_0() {
    __builtin_amdgcn_s_wait_tensorcnt(0);
    asm volatile("" ::: "memory");
}
#endif // USE_TDM

// ---------------------------------------------------------------------------
// Generic Y = X @ W^T + bias, X:[M,256] f32, W:[256,256] f32.
// One wave computes a 16(M) x 64(N) tile (4 accumulators, A reused 4x).
// outMode 0: f16 head-major [B,H,S,32]
// outMode 1: f16 transposed [B,H,32,S]   (for V)
// outMode 2: f32 flat       [M,256]      (final output)
// ---------------------------------------------------------------------------
__global__ void __launch_bounds__(32)
gemm256_wmma(const float* __restrict__ X, const float* __restrict__ W,
             const float* __restrict__ bias, void* __restrict__ Y, int outMode)
{
    const int lane = threadIdx.x & 31;
    const int g    = lane >> 4;   // half-wave group (0/1)
    const int r    = lane & 15;
    const int m0   = blockIdx.x * 16;
    const int n0   = blockIdx.y * 64;

    v8f acc[4];
    #pragma unroll
    for (int t = 0; t < 4; ++t) acc[t] = {};

    const float* xrow = X + (size_t)(m0 + r) * DM;

    for (int k0 = 0; k0 < DM; k0 += 32) {
        // A fragment: row = r, k = k0+16g+i (contiguous)
        HFrag a;
        #pragma unroll
        for (int i = 0; i < 8; ++i) {
            a.e[i]     = (_Float16)xrow[k0 + 16 * g + i];
            a.e[8 + i] = (_Float16)xrow[k0 + 16 * g + 8 + i];
        }
        #pragma unroll
        for (int t = 0; t < 4; ++t) {
            // B fragment: col n = n0+16t+r, k = k0+16g+i  (contiguous W row)
            const float* wrow = W + (size_t)(n0 + t * 16 + r) * DM + k0 + 16 * g;
            HFrag bm;
            #pragma unroll
            for (int i = 0; i < 16; ++i) bm.e[i] = (_Float16)wrow[i];
            acc[t] = wmma_f16(a.v, bm.v, acc[t]);
        }
    }

    // C layout: vgpr i, lanes[g-half] -> row = m0 + i + 8g, col = n0+16t+r
    #pragma unroll
    for (int t = 0; t < 4; ++t) {
        const int   col  = n0 + t * 16 + r;
        const float bval = bias[col];
        #pragma unroll
        for (int i = 0; i < 8; ++i) {
            const int   row = m0 + i + 8 * g;
            const float val = acc[t][i] + bval;
            if (outMode == 2) {
                ((float*)Y)[(size_t)row * DM + col] = val;
            } else {
                const int bi = row >> 11;       // row / S
                const int s  = row & (SS - 1);
                const int h  = col >> 5;        // col / HD
                const int d  = col & (HD - 1);
                if (outMode == 0) {
                    ((_Float16*)Y)[(((size_t)(bi * NH + h)) * SS + s) * HD + d] =
                        (_Float16)val;
                } else {
                    ((_Float16*)Y)[(((size_t)(bi * NH + h)) * HD + d) * SS + s] =
                        (_Float16)val;
                }
            }
        }
    }
}

// ---------------------------------------------------------------------------
// Flash attention: one wave per (b, h, 16-query tile), 32 keys per step.
// K tile: TDM tensor_load_to_lds.  V tile: global_load_async_to_lds_b128.
// Both double-buffered in LDS.
// ---------------------------------------------------------------------------
__global__ void __launch_bounds__(32)
flash_attn(const _Float16* __restrict__ Qh, const _Float16* __restrict__ Kh,
           const _Float16* __restrict__ Vt, const int* __restrict__ mask,
           float* __restrict__ Oacc)
{
    __shared__ _Float16 Kbuf[2][32 * HD];   // [key 0..31][dim 0..31]   2KB each
    __shared__ _Float16 Vbuf[2][HD * 32];   // [dim 0..31][key 0..31]   2KB each
    __shared__ _Float16 Pst[16 * 33];       // P tile staging, padded rows

    const int lane = threadIdx.x & 31;
    const int g    = lane >> 4;
    const int r    = lane & 15;
    const int qt   = blockIdx.x;        // query tile (16 rows)
    const int h    = blockIdx.y;
    const int b    = blockIdx.z;
    const int bh   = b * NH + h;

    const float scale   = 0.17677669529663687f;   // 1/sqrt(32)
    const float NEG_INF = -__builtin_inff();
    const v8f   vzero   = {};

    const _Float16* kbh   = Kh + (size_t)bh * SS * HD;
    const _Float16* vbh   = Vt + (size_t)bh * HD * SS;
    const int*      mbase = mask + (size_t)b * SS * SS;

    // Q tile as WMMA A operand (rows contiguous 32 f16 each)
    const _Float16* qbase = Qh + ((size_t)bh * SS + qt * 16) * HD;
    HFrag aq;
    aq.q[0] = *(const u32x4*)(qbase + r * HD + 16 * g);
    aq.q[1] = *(const u32x4*)(qbase + r * HD + 16 * g + 8);

    float mrun[8], lrun[8];
    #pragma unroll
    for (int i = 0; i < 8; ++i) { mrun[i] = NEG_INF; lrun[i] = 0.0f; }
    v8f o0 = {}, o1 = {};

    // -- stage one 32-key K tile + V tile into LDS buffer `bufi` --
    auto stage = [&](int j0s, int bufi) {
        const unsigned kl = (unsigned)(uintptr_t)&Kbuf[bufi][0];
        const unsigned vl = (unsigned)(uintptr_t)&Vbuf[bufi][0];
#if USE_TDM
        // K tile: rows j0s..j0s+31 of the [S, HD] f16 head matrix -> one TDM op
        tdm_load_tile32x32_f16(kl, (const void*)(kbh + (size_t)j0s * HD),
                               HD, SS, HD);
#else
        // fallback: flat 2KB async copy (tile is contiguous in Kh)
        const char* kg = (const char*)(kbh + (size_t)j0s * HD);
        #pragma unroll
        for (int i2 = 0; i2 < 4; ++i2) {
            const unsigned off = i2 * 512 + lane * 16;
            async_copy_b128(kl + off, kg + off);
        }
#endif
        // V tile: 32 dim-rows of 64B (row stride S*2B) -> [dim][key] in LDS
        const int dimq = lane >> 2;     // 0..7
        const int ch   = lane & 3;      // 16B chunk within 64B row
        #pragma unroll
        for (int i2 = 0; i2 < 4; ++i2) {
            const int dim = i2 * 8 + dimq;
            const _Float16* vg = vbh + (size_t)dim * SS + j0s + ch * 8;
            async_copy_b128(vl + (unsigned)(dim * 64 + ch * 16), (const void*)vg);
        }
    };

    stage(0, 0);   // preload tile 0

    for (int jt = 0; jt < SS / 32; ++jt) {
        const int j0  = jt * 32;
        const int cur = jt & 1;

        if (jt + 1 < SS / 32) {
            wait_dscnt_0();                 // WAR: prior ds reads of buf done
            stage(j0 + 32, cur ^ 1);        // fill next buffer
#if USE_TDM
            wait_tensor_le1();              // current K tile (oldest TDM) done
            wait_async_le4();               // current V tile (oldest 4) done
#else
            wait_async_le8();
#endif
        } else {
#if USE_TDM
            wait_tensor_0();
#endif
            wait_async_0();
        }

        if (j0 + 32 < SS) {                 // prefetch next mask rows
            __builtin_prefetch(
                (const void*)(mbase + (size_t)(qt * 16 + lane / 2) * SS + j0 + 32),
                0, 1);
        }

        // K fragments from LDS: col = key r / 16+r, k = dim 16g+i
        const _Float16* kb = &Kbuf[cur][0];
        HFrag bk0, bk1;
        bk0.q[0] = *(const u32x4*)(kb + r * HD + 16 * g);
        bk0.q[1] = *(const u32x4*)(kb + r * HD + 16 * g + 8);
        bk1.q[0] = *(const u32x4*)(kb + (16 + r) * HD + 16 * g);
        bk1.q[1] = *(const u32x4*)(kb + (16 + r) * HD + 16 * g + 8);

        v8f e0 = wmma_f16(aq.v, bk0.v, vzero);
        v8f e1 = wmma_f16(aq.v, bk1.v, vzero);

        // mask==0 -> -inf, then scale (equivalent to reference order)
        #pragma unroll
        for (int i = 0; i < 8; ++i) {
            const int  qrow = qt * 16 + i + 8 * g;
            const int* mrow = mbase + (size_t)qrow * SS + j0;
            e0[i] = mrow[r]      ? e0[i] * scale : NEG_INF;
            e1[i] = mrow[16 + r] ? e1[i] * scale : NEG_INF;
        }

        // online softmax per row (row elements live across 16 lanes of a half)
        #pragma unroll
        for (int i = 0; i < 8; ++i) {
            const float rmax = rowmax16(fmaxf(e0[i], e1[i]));
            const float mnew = fmaxf(mrun[i], rmax);
            const float corr = (mrun[i] == NEG_INF) ? 0.0f : __expf(mrun[i] - mnew);
            const float p0   = (mnew == NEG_INF) ? 0.0f : __expf(e0[i] - mnew);
            const float p1   = (mnew == NEG_INF) ? 0.0f : __expf(e1[i] - mnew);
            lrun[i] = lrun[i] * corr + rowsum16(p0 + p1);
            mrun[i] = mnew;
            o0[i] *= corr;
            o1[i] *= corr;
            // stage P (C layout -> LDS) for A-layout reload
            Pst[(i + 8 * g) * 33 + r]      = (_Float16)p0;
            Pst[(i + 8 * g) * 33 + 16 + r] = (_Float16)p1;
        }
        __syncthreads();

        HFrag pa;
        #pragma unroll
        for (int i = 0; i < 8; ++i) {
            pa.e[i]     = Pst[r * 33 + 16 * g + i];
            pa.e[8 + i] = Pst[r * 33 + 16 * g + 8 + i];
        }
        __syncthreads();

        // V fragments from LDS: col = dim r / 16+r, k = key 16g+i
        const _Float16* vb = &Vbuf[cur][0];
        HFrag bv0, bv1;
        bv0.q[0] = *(const u32x4*)(vb + r * 32 + 16 * g);
        bv0.q[1] = *(const u32x4*)(vb + r * 32 + 16 * g + 8);
        bv1.q[0] = *(const u32x4*)(vb + (16 + r) * 32 + 16 * g);
        bv1.q[1] = *(const u32x4*)(vb + (16 + r) * 32 + 16 * g + 8);

        o0 = wmma_f16(pa.v, bv0.v, o0);
        o1 = wmma_f16(pa.v, bv1.v, o1);
    }

    // epilogue: O /= l, write f32 [B,S,256] (merged heads)
    #pragma unroll
    for (int i = 0; i < 8; ++i) {
        const int   qrow = qt * 16 + i + 8 * g;
        const float inv  = 1.0f / lrun[i];
        float* orow = Oacc + ((size_t)b * SS + qrow) * DM + h * HD;
        orow[r]      = o0[i] * inv;
        orow[16 + r] = o1[i] * inv;
    }
}

// ---------------------------------------------------------------------------
extern "C" void kernel_launch(void* const* d_in, const int* in_sizes, int n_in,
                              void* d_out, int out_size, void* d_ws, size_t ws_size,
                              hipStream_t stream)
{
    (void)in_sizes; (void)n_in; (void)out_size; (void)ws_size;

    const float* query = (const float*)d_in[0];
    const float* key   = (const float*)d_in[1];
    const float* value = (const float*)d_in[2];
    const int*   mask  = (const int*)d_in[3];
    const float* Wq = (const float*)d_in[4];  const float* bq = (const float*)d_in[5];
    const float* Wk = (const float*)d_in[6];  const float* bk = (const float*)d_in[7];
    const float* Wv = (const float*)d_in[8];  const float* bv = (const float*)d_in[9];
    const float* Wo = (const float*)d_in[10]; const float* bo = (const float*)d_in[11];
    float* out = (float*)d_out;

    // Workspace carve-out (20 MB total)
    const size_t nElem = (size_t)BB * NH * SS * HD;   // 2,097,152
    _Float16* Qh = (_Float16*)d_ws;
    _Float16* Kh = Qh + nElem;
    _Float16* Vt = Kh + nElem;
    float*    Oa = (float*)(Vt + nElem);

    const int M = BB * SS;                            // 8192
    dim3 gGemm(M / 16, DM / 64);                      // 512 x 4
    dim3 gAttn(SS / 16, NH, BB);                      // 128 x 8 x 4

    gemm256_wmma<<<gGemm, 32, 0, stream>>>(query, Wq, bq, (void*)Qh, 0);
    gemm256_wmma<<<gGemm, 32, 0, stream>>>(key,   Wk, bk, (void*)Kh, 0);
    gemm256_wmma<<<gGemm, 32, 0, stream>>>(value, Wv, bv, (void*)Vt, 1);
    flash_attn  <<<gAttn, 32, 0, stream>>>(Qh, Kh, Vt, mask, Oa);
    gemm256_wmma<<<gGemm, 32, 0, stream>>>(Oa, Wo, bo, (void*)out, 2);
}
